// CRF_12317966205246
// MI455X (gfx1250) — matrix operand order repaired and checked by reading, not compiled, not run on hardware
//
#include <hip/hip_runtime.h>
#include <hip/hip_bf16.h>

#define Bsz 256
#define Ssz 1024
#define Tsz 96
#define START_TAG 94
#define STOP_TAG  95

typedef _Float16 v16h __attribute__((ext_vector_type(16)));
typedef float    v8f  __attribute__((ext_vector_type(8)));

// A-matrix (16x32 f16) per-lane half-index -> K   (ISA 7.12.2)
__device__ __forceinline__ int kmapA(int idx, int lg) {
    return idx + 8 * lg + ((idx >= 8) ? 8 : 0);
}
// B-matrix (32x16 f16) per-lane half-index -> K   (same order as kmapA)
__device__ __forceinline__ int kmapB(int idx, int lg) {
    return idx + 8 * lg + ((idx >= 8) ? 8 : 0);
}
// bare v_log_f32 (log2) * ln2 -- no denorm/range guards needed here (x > 0.5)
__device__ __forceinline__ float fast_log(float x) {
    return __builtin_amdgcn_logf(x) * 0.69314718055994530942f;
}

// ---------------------------------------------------------------------------
// Forward algorithm. One workgroup = 16 batches, 6 waves (one N-tile each).
// part state lives in registers in WMMA C-fragment layout:
//   partReg[r] holds (batch row M = r + 8*laneGrp, tag column j = 16*wave + lane%16)
// Per step: row-max via shuffles, p=exp(part-m) -> LDS f16, q = p @ E via
// 3x v_wmma_f32_16x16x32_f16 (E register-resident), part' = ft + m + log(q).
// ---------------------------------------------------------------------------
__global__ __launch_bounds__(192) void crf_forward_kernel(
    const float* __restrict__ feats,          // (B,S,T) f32
    const unsigned char* __restrict__ mask,   // (B,S)   bool
    const float* __restrict__ trans,          // (T,T)   f32
    float* __restrict__ logZ)                 // (B,)    f32 out (workspace)
{
    __shared__ __align__(16) _Float16 pH[16][96];   // p in A-matrix source layout
    __shared__ float wmax[16][8];                   // per-wave row-max partials
    __shared__ float mrow[16];
    __shared__ int   lenLds[16];
    __shared__ float tstop[Tsz];
    __shared__ __align__(16) float finPart[16][97]; // final part for logZ

    const int tid  = threadIdx.x;
    const int w    = tid >> 5;        // wave 0..5 -> N-tile
    const int lane = tid & 31;
    const int lg   = lane >> 4;       // lane group
    const int lm   = lane & 15;
    const int b0   = blockIdx.x * 16; // batch tile base
    const int j0   = w * 16;          // N-tile base
    const int j    = j0 + lm;         // this lane's tag column

    // --- B fragments of E = exp(transitions): VGPR-resident for all steps ---
    v16h Bf[3];
    #pragma unroll
    for (int kc = 0; kc < 3; ++kc) {
        #pragma unroll
        for (int idx = 0; idx < 16; ++idx) {
            const int k = kc * 32 + kmapB(idx, lg);
            Bf[kc][idx] = (_Float16)__expf(trans[k * Tsz + j]);
        }
    }

    // --- lengths (mask is a prefix mask), stop-transition column ---
    if (tid < 16) {
        int cnt = 0;
        const unsigned char* mr = mask + (size_t)(b0 + tid) * Ssz;
        for (int t = 0; t < Ssz; ++t) cnt += mr[t];
        lenLds[tid] = cnt;
    }
    for (int jj = tid; jj < Tsz; jj += 192) tstop[jj] = trans[jj * Tsz + STOP_TAG];

    // --- init part0 = feats[:,0,:] + trans[START,:], in registers ---
    const float tstart = trans[START_TAG * Tsz + j];
    float partReg[8];
    #pragma unroll
    for (int r = 0; r < 8; ++r) {
        const int Mrow = r + 8 * lg;
        partReg[r] = feats[((size_t)(b0 + Mrow) * Ssz) * Tsz + j] + tstart;
    }
    __syncthreads();
    int lenR[8];
    #pragma unroll
    for (int r = 0; r < 8; ++r) lenR[r] = lenLds[r + 8 * lg];

    for (int t = 1; t < Ssz; ++t) {
        // prefetch f_t for this lane's 8 (row, column) cells (s_clause'd, 64B/group)
        float ftv[8];
        #pragma unroll
        for (int r = 0; r < 8; ++r)
            ftv[r] = feats[(((size_t)(b0 + r + 8 * lg) * Ssz) + t) * Tsz + j];

        // stage A: row max.  Row is constant across each 16-lane group ->
        // 4 shfl_xor steps, then one LDS partial per wave, tiny cross-wave combine.
        #pragma unroll
        for (int r = 0; r < 8; ++r) {
            float v = partReg[r];
            #pragma unroll
            for (int s = 1; s < 16; s <<= 1)
                v = fmaxf(v, __shfl_xor(v, s, 32));
            if (lm == 0) wmax[r + 8 * lg][w] = v;
        }
        __syncthreads();                              // barrier 1
        if (tid < 16) {
            float mx = wmax[tid][0];
            #pragma unroll
            for (int q = 1; q < 6; ++q) mx = fmaxf(mx, wmax[tid][q]);
            mrow[tid] = mx;
        }
        __syncthreads();                              // barrier 2

        // stage B: p = exp(part - m) -> LDS f16 (A-matrix source layout)
        float mR[8];
        #pragma unroll
        for (int r = 0; r < 8; ++r) {
            mR[r] = mrow[r + 8 * lg];
            pH[r + 8 * lg][j] = (_Float16)__expf(partReg[r] - mR[r]);
        }
        __syncthreads();                              // barrier 3

        // stage C: gather ALL three A fragments first (6x ds_load_b128 in one
        // clause, single dscnt wait), then the serialized accumulator chain.
        v16h a0, a1, a2;
        #pragma unroll
        for (int idx = 0; idx < 16; ++idx) a0[idx] = pH[lm][ 0 + kmapA(idx, lg)];
        #pragma unroll
        for (int idx = 0; idx < 16; ++idx) a1[idx] = pH[lm][32 + kmapA(idx, lg)];
        #pragma unroll
        for (int idx = 0; idx < 16; ++idx) a2[idx] = pH[lm][64 + kmapA(idx, lg)];

        v8f acc = {0.f, 0.f, 0.f, 0.f, 0.f, 0.f, 0.f, 0.f};
        acc = __builtin_amdgcn_wmma_f32_16x16x32_f16(false, a0, false, Bf[0],
                                                     (short)0, acc, false, false);
        acc = __builtin_amdgcn_wmma_f32_16x16x32_f16(false, a1, false, Bf[1],
                                                     (short)0, acc, false, false);
        acc = __builtin_amdgcn_wmma_f32_16x16x32_f16(false, a2, false, Bf[2],
                                                     (short)0, acc, false, false);

        // stage D: branchless masked update (v_cndmask, bare v_log_f32).
        // No loop-end barrier needed: every wave's pH reads complete (dscnt wait)
        // before it signals barrier 1 of the next step, and pH writers sit behind
        // barrier 2 of the next step.
        #pragma unroll
        for (int r = 0; r < 8; ++r) {
            const float nv = ftv[r] + mR[r] + fast_log(acc[r]);
            partReg[r] = (t < lenR[r]) ? nv : partReg[r];
        }
    }

    // --- finalize: logZ[b] = logsumexp_i(part[b,i] + trans[i,STOP]) ---
    #pragma unroll
    for (int r = 0; r < 8; ++r) finPart[r + 8 * lg][j] = partReg[r];
    __syncthreads();
    if (tid < 16) {
        float mx = -3.4e38f;
        for (int i = 0; i < Tsz; ++i) mx = fmaxf(mx, finPart[tid][i] + tstop[i]);
        float s = 0.f;
        for (int i = 0; i < Tsz; ++i) s += __expf(finPart[tid][i] + tstop[i] - mx);
        logZ[b0 + tid] = mx + fast_log(s);
    }
}

// ---------------------------------------------------------------------------
// Gold score: one block per batch; deterministic fixed-order tree reduction.
// ---------------------------------------------------------------------------
__global__ __launch_bounds__(128) void crf_gold_kernel(
    const float* __restrict__ feats,
    const unsigned char* __restrict__ mask,
    const long long* __restrict__ y,          // int64
    const float* __restrict__ trans,
    float* __restrict__ gold)                 // (B,) out (workspace)
{
    __shared__ float sred[128];
    __shared__ int   cred[128];
    const int b = blockIdx.x, tid = threadIdx.x;
    float acc = 0.f; int cnt = 0;
    for (int t = tid; t < Ssz; t += 128) {
        if (mask[(size_t)b * Ssz + t]) {
            const int yt   = (int)y[(size_t)b * Ssz + t];
            const int prev = (t == 0) ? START_TAG : (int)y[(size_t)b * Ssz + t - 1];
            acc += feats[((size_t)b * Ssz + t) * Tsz + yt] + trans[prev * Tsz + yt];
            ++cnt;
        }
    }
    sred[tid] = acc; cred[tid] = cnt;
    __syncthreads();
    for (int off = 64; off > 0; off >>= 1) {
        if (tid < off) { sred[tid] += sred[tid + off]; cred[tid] += cred[tid + off]; }
        __syncthreads();
    }
    if (tid == 0) {
        const int len  = cred[0];
        const int last = (int)y[(size_t)b * Ssz + len - 1];
        gold[b] = sred[0] + trans[last * Tsz + STOP_TAG];
    }
}

// ---------------------------------------------------------------------------
// Final scalar: sum(logZ) - sum(gold), fixed-order tree -> deterministic.
// ---------------------------------------------------------------------------
__global__ __launch_bounds__(256) void crf_reduce_kernel(
    const float* __restrict__ ws, float* __restrict__ out)
{
    __shared__ float sred[256];
    const int tid = threadIdx.x;
    sred[tid] = ws[tid] - ws[Bsz + tid];
    __syncthreads();
    for (int off = 128; off > 0; off >>= 1) {
        if (tid < off) sred[tid] += sred[tid + off];
        __syncthreads();
    }
    if (tid == 0) out[0] = sred[0];
}

extern "C" void kernel_launch(void* const* d_in, const int* in_sizes, int n_in,
                              void* d_out, int out_size, void* d_ws, size_t ws_size,
                              hipStream_t stream) {
    const float*         feats = (const float*)d_in[0];          // (B,S,T) f32
    const unsigned char* mask  = (const unsigned char*)d_in[1];  // (B,S) bool
    const long long*     y     = (const long long*)d_in[2];      // (B,S) int64
    const float*         trans = (const float*)d_in[3];          // (T,T) f32

    float* ws   = (float*)d_ws;
    float* logZ = ws;           // [0,256)
    float* gold = ws + Bsz;     // [256,512)
    float* out  = (float*)d_out;

    crf_forward_kernel<<<Bsz / 16, 192, 0, stream>>>(feats, mask, trans, logZ);
    crf_gold_kernel<<<Bsz, 128, 0, stream>>>(feats, mask, y, trans, gold);
    crf_reduce_kernel<<<1, 256, 0, stream>>>(ws, out);
}
